// H2OLlamaAttention_streaming_47923245089257
// MI455X (gfx1250) — matrix operand order repaired and checked by reading, not compiled, not run on hardware
//
#include <hip/hip_runtime.h>
#include <stdint.h>

// ---------------------------------------------------------------------------
// MI455X / gfx1250 implementation of H2O-Llama attention layer.
// Compute-bound (AI ~1000 FLOP/B) -> all matmuls via v_wmma_f32_16x16x32_bf16.
// GEMM uses ASYNCcnt-tracked global->LDS async copies (double buffered) so the
// tensor-DMA path overlaps WMMA issue. wave32 everywhere; fragment layouts per
// CDNA5 ISA §7.12.2.
// ---------------------------------------------------------------------------

typedef __attribute__((ext_vector_type(16))) __bf16 bf16x16;
typedef __attribute__((ext_vector_type(8)))  float  f32x8;
typedef __attribute__((ext_vector_type(4)))  int    i32x4;

union Frag {
  bf16x16 v;
  uint4 u[2];
  unsigned short s[16];
};

__device__ inline unsigned short f2bf(float f) {
  union { float f; unsigned u; } c; c.f = f;
  unsigned u = c.u;
  // round-to-nearest-even f32 -> bf16
  return (unsigned short)((u + 0x7FFFu + ((u >> 16) & 1u)) >> 16);
}

// ---- async global->LDS copy (16B per lane), tracked by ASYNCcnt ------------
__device__ inline void async_copy_b128(const unsigned short* g, unsigned short* l) {
#if __has_builtin(__builtin_amdgcn_global_load_async_to_lds_b128)
  // Builtin signature (from hipcc diagnostic): (AS1 int4* src, AS3 int4* dst,
  // int offset, int cpol). Integer round-trip drops constness; low 32 bits of
  // a generic shared pointer are the LDS byte address (aperture rules).
  __builtin_amdgcn_global_load_async_to_lds_b128(
      (__attribute__((address_space(1))) i32x4*)(unsigned long long)g,
      (__attribute__((address_space(3))) i32x4*)(unsigned)(unsigned long long)l,
      0, 0);
#else
  unsigned lds_off = (unsigned)(unsigned long long)l;
  asm volatile("global_load_async_to_lds_b128 %0, %1, off"
               :: "v"(lds_off), "v"(g)
               : "memory");
#endif
}

__device__ inline void async_wait_6() {
#if __has_builtin(__builtin_amdgcn_s_wait_asynccnt)
  __builtin_amdgcn_s_wait_asynccnt(6);
#else
  asm volatile("s_wait_asynccnt 0x6" ::: "memory");
#endif
}
__device__ inline void async_wait_0() {
#if __has_builtin(__builtin_amdgcn_s_wait_asynccnt)
  __builtin_amdgcn_s_wait_asynccnt(0);
#else
  asm volatile("s_wait_asynccnt 0x0" ::: "memory");
#endif
}

// A-matrix 16x32 bf16 fragment (M x K), row-major source with leading dim ld.
// Lanes 0-15: row = m0+lane, elems 0..7 = K[kk..kk+7], elems 8..15 = K[kk+16..kk+23]
// Lanes 16-31: row = m0+lane-16, K offset +8 (per ISA 16-bit A layout).
__device__ inline Frag loadA16(const unsigned short* __restrict__ base, int ld,
                               int m0, int kk, int lane) {
  Frag f;
  const unsigned short* p = base + (m0 + (lane & 15)) * ld + kk + ((lane >> 4) << 3);
  f.u[0] = *reinterpret_cast<const uint4*>(p);
  f.u[1] = *reinterpret_cast<const uint4*>(p + 16);
  return f;
}

// B-matrix 32x16 bf16 fragment (K x N) where source is row-major [N][K]
// (i.e. B[k][n] = src[n][k], exactly the Linear-weight / K-matrix case).
// Lane n (0-15) holds column n, elems 0..7 = K[kk..kk+7], elems 8..15 = K[kk+16..kk+23]; lanes 16-31 K offset +8.
__device__ inline Frag loadB16(const unsigned short* __restrict__ base, int ld,
                               int n0, int kk, int lane) {
  Frag f;
  const unsigned short* p = base + (n0 + (lane & 15)) * ld + kk + ((lane >> 4) << 3);
  f.u[0] = *reinterpret_cast<const uint4*>(p);
  f.u[1] = *reinterpret_cast<const uint4*>(p + 16);
  return f;
}

// ---------------------------------------------------------------------------
// f32 -> bf16 conversion (once per tensor; keeps cvt VALU off the WMMA loop)
// ---------------------------------------------------------------------------
__global__ __launch_bounds__(256) void cvt_f32_bf16(const float* __restrict__ in,
                                                    unsigned short* __restrict__ out,
                                                    int n) {
  int i = blockIdx.x * blockDim.x + threadIdx.x;
  if (i < n) out[i] = f2bf(in[i]);
}

// ---------------------------------------------------------------------------
// C[M,N] f32 = A_bf16[M,K] @ B_bf16[N,K]^T    (torch Linear: x @ W^T)
// Block (4 waves) computes a 64x128 tile; each wave a 32x64 quadrant
// (2x4 WMMA accumulators). Per 32-deep k-step the block async-copies
// A(64x32)+B(128x32) bf16 tiles (12KB) into a double-buffered LDS stage
// (6 b128 copies per thread), overlapping the next stage with WMMAs.
// ---------------------------------------------------------------------------
__global__ __launch_bounds__(128) void gemm_bf16_nt(const unsigned short* __restrict__ A,
                                                    const unsigned short* __restrict__ B,
                                                    float* __restrict__ C,
                                                    int M, int N, int K) {
  __shared__ __align__(16) unsigned short sA[2][64 * 32];    //  8 KB
  __shared__ __align__(16) unsigned short sB[2][128 * 32];   // 16 KB

  const int tid  = threadIdx.x;
  const int lane = tid & 31;
  const int wv   = tid >> 5;
  const int wm   = (wv >> 1) << 5;       // 0 / 32
  const int wn   = (wv & 1) << 6;        // 0 / 64
  const int tilesN = N >> 7;
  const int m0 = (blockIdx.x / tilesN) << 6;
  const int n0 = (blockIdx.x % tilesN) << 7;

  // chunk descriptors: A has 256 16B chunks (2/thread), B has 512 (4/thread)
  const int rowA0 = (tid) >> 2,         colA0 = (tid & 3) << 3;
  const int rowA1 = (tid + 128) >> 2,   colA1 = ((tid + 128) & 3) << 3;

  auto issue = [&](int kk, int buf) {
    async_copy_b128(A + (m0 + rowA0) * K + kk + colA0, &sA[buf][rowA0 * 32 + colA0]);
    async_copy_b128(A + (m0 + rowA1) * K + kk + colA1, &sA[buf][rowA1 * 32 + colA1]);
#pragma unroll
    for (int q = 0; q < 4; ++q) {
      int c   = tid + (q << 7);
      int row = c >> 2, col = (c & 3) << 3;
      async_copy_b128(B + (n0 + row) * K + kk + col, &sB[buf][row * 32 + col]);
    }
  };

  f32x8 acc[2][4] = {};
  const int nk = K >> 5;

  issue(0, 0);
  for (int i = 0; i < nk; ++i) {
    const int buf = i & 1;
    if (i + 1 < nk) { issue((i + 1) << 5, buf ^ 1); async_wait_6(); }
    else            { async_wait_0(); }
    __syncthreads();                      // stage i visible to all 4 waves

    Frag a0 = loadA16(sA[buf], 32, wm,      0, lane);
    Frag a1 = loadA16(sA[buf], 32, wm + 16, 0, lane);
#pragma unroll
    for (int j = 0; j < 4; ++j) {
      Frag b = loadB16(sB[buf], 32, wn + j * 16, 0, lane);
      acc[0][j] = __builtin_amdgcn_wmma_f32_16x16x32_bf16(
          false, a0.v, false, b.v, (short)0, acc[0][j], false, false);
      acc[1][j] = __builtin_amdgcn_wmma_f32_16x16x32_bf16(
          false, a1.v, false, b.v, (short)0, acc[1][j], false, false);
    }
    __syncthreads();                      // reads done before buf is re-filled
  }

  // C/D layout: VGPR r -> row r (lanes 0-15) / row r+8 (lanes 16-31); col = lane&15
  int rb = (lane >> 4) << 3;
  int cl = lane & 15;
#pragma unroll
  for (int i = 0; i < 2; ++i)
#pragma unroll
    for (int j = 0; j < 4; ++j)
#pragma unroll
      for (int r = 0; r < 8; ++r)
        C[(m0 + wm + i * 16 + rb + r) * N + (n0 + wn + j * 16 + cl)] = acc[i][j][r];
}

// ---------------------------------------------------------------------------
// RoPE on Q,K; emit head-major bf16 Q/K for the attention WMMAs and keep
// roped f32 K (layout [S,H]) for the eviction gather.
// ---------------------------------------------------------------------------
__global__ __launch_bounds__(256) void rope_qk(const float* __restrict__ Qf,
                                               float* __restrict__ Kf,
                                               unsigned short* __restrict__ Qb,
                                               unsigned short* __restrict__ Kb) {
  const int S = 2048, H = 4096;
  int idx = blockIdx.x * blockDim.x + threadIdx.x;   // NH * S * 64
  if (idx >= 32 * S * 64) return;
  int h   = idx / (S * 64);
  int rem = idx - h * S * 64;
  int s   = rem >> 6;
  int d   = rem & 63;
  // inv_freq = 10000^(-d/64) = exp(-d * ln(1e4)/64)
  float inv = __expf(-(float)d * 0.14391156516f);
  float fr  = (float)s * inv;
  float sn, cs;
  __sincosf(fr, &sn, &cs);
  int src = s * H + h * 128 + d;
  float q1 = Qf[src], q2 = Qf[src + 64];
  float k1 = Kf[src], k2 = Kf[src + 64];
  float qo1 = q1 * cs - q2 * sn, qo2 = q2 * cs + q1 * sn;
  float ko1 = k1 * cs - k2 * sn, ko2 = k2 * cs + k1 * sn;
  int dst = (h * S + s) * 128 + d;                   // [NH][S][HD]
  Qb[dst] = f2bf(qo1); Qb[dst + 64] = f2bf(qo2);
  Kb[dst] = f2bf(ko1); Kb[dst + 64] = f2bf(ko2);
  Kf[src] = ko1; Kf[src + 64] = ko2;                 // roped f32 K for eviction
}

// V transposed to [NH][HD][S] bf16 so the P@V B-fragment is a contiguous load.
__global__ __launch_bounds__(256) void v_transpose(const float* __restrict__ Vf,
                                                   unsigned short* __restrict__ Vt) {
  const int S = 2048, H = 4096, HD = 128;
  int idx = blockIdx.x * blockDim.x + threadIdx.x;   // NH * S * HD
  if (idx >= 32 * S * HD) return;
  int h   = idx / (S * HD);
  int rem = idx - h * S * HD;
  int s   = rem >> 7;
  int d   = rem & 127;
  Vt[(h * HD + d) * S + s] = f2bf(Vf[s * H + h * 128 + d]);
}

// ---------------------------------------------------------------------------
// Flash attention: one wave (single-wave workgroup -> barriers are S_NOP)
// per (head, 16-query tile). 32-key blocks: 8 score WMMAs + online softmax
// through LDS + 8 P@V WMMAs. Causal: skip fully-masked blocks.
// ---------------------------------------------------------------------------
__global__ __launch_bounds__(32) void flash_attn(const unsigned short* __restrict__ Qb,
                                                 const unsigned short* __restrict__ Kb,
                                                 const unsigned short* __restrict__ Vt,
                                                 unsigned short* __restrict__ ctx) {
  const int S = 2048, HD = 128, H = 4096;
  const float SCALE = 0.08838834764831845f;          // 1/sqrt(128)
  int h    = blockIdx.x >> 7;
  int m0   = (blockIdx.x & 127) << 4;
  int lane = threadIdx.x;
  const unsigned short* Qh = Qb + h * S * HD;
  const unsigned short* Kh = Kb + h * S * HD;
  const unsigned short* Vh = Vt + h * HD * S;

  __shared__ float sP[16][33];                       // padded f32 scores
  __shared__ __align__(16) unsigned short sPb[16][32];  // bf16 P, A-layout source
  __shared__ float sRow[16];                         // alpha / 1/l broadcast

  Frag aQ[4];
#pragma unroll
  for (int c = 0; c < 4; ++c) aQ[c] = loadA16(Qh, HD, m0, c * 32, lane);

  f32x8 O[8] = {};
  float mrow = -3.0e38f, lrow = 0.0f;                // valid in lanes 0-15
  int rb = (lane >> 4) << 3;
  int cl = lane & 15;
  int nJ = (m0 + 15) / 32 + 1;                       // causal block count

  for (int j = 0; j < nJ; ++j) {
    int kb0 = j * 32;
    f32x8 s0 = {}, s1 = {};
#pragma unroll
    for (int c = 0; c < 4; ++c) {
      Frag bk0 = loadB16(Kh, HD, kb0,      c * 32, lane);
      Frag bk1 = loadB16(Kh, HD, kb0 + 16, c * 32, lane);
      s0 = __builtin_amdgcn_wmma_f32_16x16x32_bf16(false, aQ[c].v, false, bk0.v,
                                                   (short)0, s0, false, false);
      s1 = __builtin_amdgcn_wmma_f32_16x16x32_bf16(false, aQ[c].v, false, bk1.v,
                                                   (short)0, s1, false, false);
    }
    // scale + causal mask, stage to LDS [row][key]
#pragma unroll
    for (int r = 0; r < 8; ++r) {
      int rl = rb + r, grow = m0 + rl;
      float v0 = s0[r] * SCALE; if (kb0 + cl      > grow) v0 = -3.0e38f;
      float v1 = s1[r] * SCALE; if (kb0 + 16 + cl > grow) v1 = -3.0e38f;
      sP[rl][cl] = v0; sP[rl][cl + 16] = v1;
    }
    __syncthreads();
    if (lane < 16) {                                 // per-row online softmax
      float bm = -3.0e38f;
#pragma unroll
      for (int c = 0; c < 32; ++c) bm = fmaxf(bm, sP[lane][c]);
      float mnew  = fmaxf(mrow, bm);
      float alpha = __expf(mrow - mnew);
      float ls = 0.0f;
#pragma unroll
      for (int c = 0; c < 32; ++c) {
        float p = __expf(sP[lane][c] - mnew);
        sPb[lane][c] = f2bf(p);
        ls += p;
      }
      lrow = lrow * alpha + ls;
      mrow = mnew;
      sRow[lane] = alpha;
    }
    __syncthreads();
    float al[8];
#pragma unroll
    for (int r = 0; r < 8; ++r) al[r] = sRow[rb + r];
#pragma unroll
    for (int t = 0; t < 8; ++t)
#pragma unroll
      for (int r = 0; r < 8; ++r) O[t][r] = O[t][r] * al[r];

    // P (16x32) as A-fragment straight from LDS (bf16, A layout addressing)
    Frag aP;
    {
      const unsigned short* p = &sPb[lane & 15][(lane >> 4) << 3];
      aP.u[0] = *reinterpret_cast<const uint4*>(p);
      aP.u[1] = *reinterpret_cast<const uint4*>(p + 16);
    }
#pragma unroll
    for (int t = 0; t < 8; ++t) {
      Frag bv = loadB16(Vh, S, t * 16, kb0, lane);   // V^T[dim][key]
      O[t] = __builtin_amdgcn_wmma_f32_16x16x32_bf16(false, aP.v, false, bv.v,
                                                     (short)0, O[t], false, false);
    }
    __syncthreads();
  }

  if (lane < 16) sRow[lane] = 1.0f / lrow;
  __syncthreads();
  float li[8];
#pragma unroll
  for (int r = 0; r < 8; ++r) li[r] = sRow[rb + r];
#pragma unroll
  for (int t = 0; t < 8; ++t)
#pragma unroll
    for (int r = 0; r < 8; ++r) {
      int row = m0 + rb + r;
      int col = h * HD + t * 16 + cl;
      ctx[row * H + col] = f2bf(O[t][r] * li[r]);    // ctx bf16 [S][H]
    }
}

// ---------------------------------------------------------------------------
// H2O eviction gather: sink(4) + recent(204) tokens per head, K and V.
// ---------------------------------------------------------------------------
__global__ __launch_bounds__(256) void evict_kv(const float* __restrict__ Kf,
                                                const float* __restrict__ Vf,
                                                float* __restrict__ outK,
                                                float* __restrict__ outV) {
  const int H = 4096, PER = 208 * 128, TOTAL = 32 * PER;
  int idx = blockIdx.x * blockDim.x + threadIdx.x;
  if (idx >= 2 * TOTAL) return;
  int which = idx / TOTAL;
  int i     = idx - which * TOTAL;
  int h     = i / PER;
  int rem   = i - h * PER;
  int t     = rem >> 7;
  int d     = rem & 127;
  int s     = (t < 4) ? t : (1844 + (t - 4));        // 2048-204 = 1844
  float val = (which ? Vf : Kf)[s * H + h * 128 + d];
  (which ? outV : outK)[i] = val;
}

// ---------------------------------------------------------------------------
extern "C" void kernel_launch(void* const* d_in, const int* in_sizes, int n_in,
                              void* d_out, int out_size, void* d_ws, size_t ws_size,
                              hipStream_t stream) {
  (void)in_sizes; (void)n_in; (void)out_size; (void)ws_size;
  const int S = 2048, H = 4096;
  const float* X  = (const float*)d_in[0];
  const float* Wq = (const float*)d_in[1];
  const float* Wk = (const float*)d_in[2];
  const float* Wv = (const float*)d_in[3];
  const float* Wo = (const float*)d_in[4];
  // position_ids (d_in[5]) == arange(S); positions computed in-kernel.

  char* ws = (char*)d_ws;
  size_t o = 0;
  unsigned short* Xb  = (unsigned short*)(ws + o); o += (size_t)S * H * 2;   // 16MB
  unsigned short* Wb  = (unsigned short*)(ws + o); o += (size_t)H * H * 2;   // 32MB
  float*          Qf  = (float*)(ws + o);          o += (size_t)S * H * 4;   // 32MB
  float*          Kf  = (float*)(ws + o);          o += (size_t)S * H * 4;   // 32MB
  float*          Vf  = (float*)(ws + o);          o += (size_t)S * H * 4;   // 32MB
  unsigned short* Qb  = (unsigned short*)(ws + o); o += (size_t)S * H * 2;   // 16MB
  unsigned short* Kb  = (unsigned short*)(ws + o); o += (size_t)S * H * 2;   // 16MB
  unsigned short* Vt  = (unsigned short*)(ws + o); o += (size_t)S * H * 2;   // 16MB
  unsigned short* Ctx = Xb;  // Xb dead after QKV GEMMs; reuse for ctx bf16

  float* outAttn = (float*)d_out;
  float* outK    = outAttn + (size_t)S * H;          // 8,388,608
  float* outV    = outK + 32 * 208 * 128;            // + 851,968

  const int nX = S * H;        // 8,388,608
  const int nW = H * H;        // 16,777,216
  const int gemmBlocks = (S / 64) * (H / 128);       // 32 x 32 = 1024 blocks

  // 1) activations -> bf16
  cvt_f32_bf16<<<nX / 256, 256, 0, stream>>>(X, Xb, nX);

  // 2) Q/K/V projections (weight staged to bf16, then async-LDS WMMA GEMM)
  cvt_f32_bf16<<<nW / 256, 256, 0, stream>>>(Wq, Wb, nW);
  gemm_bf16_nt<<<gemmBlocks, 128, 0, stream>>>(Xb, Wb, Qf, S, H, H);
  cvt_f32_bf16<<<nW / 256, 256, 0, stream>>>(Wk, Wb, nW);
  gemm_bf16_nt<<<gemmBlocks, 128, 0, stream>>>(Xb, Wb, Kf, S, H, H);
  cvt_f32_bf16<<<nW / 256, 256, 0, stream>>>(Wv, Wb, nW);
  gemm_bf16_nt<<<gemmBlocks, 128, 0, stream>>>(Xb, Wb, Vf, S, H, H);

  // 3) RoPE + head-major bf16 repack; V transpose
  rope_qk<<<(32 * S * 64) / 256, 256, 0, stream>>>(Qf, Kf, Qb, Kb);
  v_transpose<<<(32 * S * 128) / 256, 256, 0, stream>>>(Vf, Vt);

  // 4) flash attention (32 heads x 128 query tiles)
  flash_attn<<<32 * 128, 32, 0, stream>>>(Qb, Kb, Vt, Ctx);

  // 5) output projection
  cvt_f32_bf16<<<nW / 256, 256, 0, stream>>>(Wo, Wb, nW);
  gemm_bf16_nt<<<gemmBlocks, 128, 0, stream>>>(Ctx, Wb, outAttn, S, H, H);

  // 6) H2O KV eviction gather
  int nEv = 2 * 32 * 208 * 128;
  evict_kv<<<(nEv + 255) / 256, 256, 0, stream>>>(Kf, Vf, outK, outV);
}
